// EfficientAttention_18717467476229
// MI455X (gfx1250) — matrix operand order repaired and checked by reading, not compiled, not run on hardware
//
#include <hip/hip_runtime.h>

// ---------------------------------------------------------------------------
// MI455X (gfx1250) attention layer: f16 WMMA (v_wmma_f32_16x16x32_f16) for all
// matmuls, f32 accumulation, async global->LDS tile staging (ASYNCcnt),
// double-buffered LDS, RoPE fused into QKV epilogue, flash attention with
// online softmax. B=2, S=2048, H=1024, NH=16, HD=64
// ---------------------------------------------------------------------------

typedef __attribute__((ext_vector_type(16))) _Float16 v16h;
typedef __attribute__((ext_vector_type(8)))  _Float16 v8h;
typedef __attribute__((ext_vector_type(4)))  _Float16 v4h;
typedef __attribute__((ext_vector_type(8)))  float    v8f;

#define B_   2
#define S_   2048
#define H_   1024
#define NH_  16
#define HD_  64
#define M_   (B_*S_)   // 4096
#define LD_  72        // LDS row stride in halves (64 + 8 pad, 144B)

static __device__ __forceinline__ v16h cat8(v8h lo, v8h hi) {
  v16h r;
#pragma unroll
  for (int i = 0; i < 8; ++i) { r[i] = lo[i]; r[i + 8] = hi[i]; }
  return r;
}

static __device__ __forceinline__ v8f wmma_f16(v16h a, v16h b, v8f c) {
  return __builtin_amdgcn_wmma_f32_16x16x32_f16(false, a, false, b,
                                                (short)0, c, false, false);
}

// A-fragment (16x32 f16) from row-major [m][k] (ISA 7.12.2 layout):
// lane L -> m = L&15, kb = (L>>4)*8; halves at row+kb and row+kb+16.
static __device__ __forceinline__ v16h load_a_frag(const _Float16* base, int ld, int lane) {
  const _Float16* row = base + (size_t)(lane & 15) * ld + ((lane >> 4) * 8);
  return cat8(*(const v8h*)row, *(const v8h*)(row + 16));
}

// B-fragment (32x16) where the source holds B transposed: T[n][k] row-major.
// lane L -> n = L&15, kb = (L>>4)*8; halves at row+kb and row+kb+16.
static __device__ __forceinline__ v16h load_bT_frag(const _Float16* base, int ld, int lane) {
  const _Float16* row = base + (size_t)(lane & 15) * ld + ((lane >> 4) * 8);
  return cat8(*(const v8h*)row, *(const v8h*)(row + 16));
}

// CDNA5 async copy: 16B global -> LDS, tracked by ASYNCcnt (no VGPR staging).
// lds_off: LDS byte offset (low 32 bits of the generic pointer == LDS offset).
static __device__ __forceinline__ void async_load_b128(unsigned lds_off, const void* gaddr) {
  asm volatile("global_load_async_to_lds_b128 %0, %1, off"
               :: "v"(lds_off), "v"(gaddr) : "memory");
}
static __device__ __forceinline__ void wait_async_12() {
  asm volatile("s_wait_asynccnt 0xc" ::: "memory");
}
static __device__ __forceinline__ void wait_async_4() {
  asm volatile("s_wait_asynccnt 0x4" ::: "memory");
}
static __device__ __forceinline__ void wait_async_0() {
  asm volatile("s_wait_asynccnt 0x0" ::: "memory");
}

// ---------------------------------------------------------------------------
// Kernel 0: f32 -> f16 cast (vectorized x4)
// ---------------------------------------------------------------------------
__global__ void cast_f32_to_f16(const float* __restrict__ src,
                                _Float16* __restrict__ dst, int n4) {
  int i = blockIdx.x * blockDim.x + threadIdx.x;
  if (i >= n4) return;
  float4 v = ((const float4*)src)[i];
  v4h o;
  o[0] = (_Float16)v.x; o[1] = (_Float16)v.y;
  o[2] = (_Float16)v.z; o[3] = (_Float16)v.w;
  ((v4h*)dst)[i] = o;
}

// ---------------------------------------------------------------------------
// Kernel 1: fused QKV projection + bias + RoPE, output [B,NH,S,HD] f16.
// Block = 128 thr (4 waves). Block tile 128(M)x64(N); K-step 64; each wave
// owns 32x64 (acc[2][4]) so every B fragment is reused by two A fragments.
// Double-buffered LDS fed by async-to-LDS loads. z = 0/1/2 -> Q/K/V.
// ---------------------------------------------------------------------------
__global__ __launch_bounds__(128)
void qkv_rope_kernel(const _Float16* __restrict__ hs,
                     const _Float16* __restrict__ Wq,
                     const _Float16* __restrict__ Wk,
                     const _Float16* __restrict__ Wv,
                     const float* __restrict__ bq,
                     const float* __restrict__ bk,
                     const float* __restrict__ bv,
                     _Float16* __restrict__ Qo,
                     _Float16* __restrict__ Ko,
                     _Float16* __restrict__ Vo) {
  const int z = blockIdx.z;
  const _Float16* W  = (z == 0) ? Wq : (z == 1) ? Wk : Wv;
  const float* bias  = (z == 0) ? bq : (z == 1) ? bk : bv;
  _Float16* dst      = (z == 0) ? Qo : (z == 1) ? Ko : Vo;

  const int tid = threadIdx.x;
  const int lane = tid & 31;
  const int wave = tid >> 5;
  const int n0 = blockIdx.x * 64;
  const int m0 = blockIdx.y * 128;

  __shared__ __align__(32) _Float16 As[2][128 * LD_];
  __shared__ __align__(32) _Float16 Ws[2][64 * LD_];

  v8f acc[2][4];
#pragma unroll
  for (int g = 0; g < 2; ++g)
#pragma unroll
    for (int j = 0; j < 4; ++j)
#pragma unroll
      for (int r = 0; r < 8; ++r) acc[g][j][r] = 0.f;

  // A: 128x64 halves = 1024 x 16B chunks (8/thread);  W: 64x64 (4/thread)
  auto issue = [&](int buf, int kk) {
#pragma unroll
    for (int c = 0; c < 8; ++c) {
      int chunk = tid + c * 128;
      int row = chunk >> 3;
      int col = (chunk & 7) * 8;
      async_load_b128((unsigned)(size_t)&As[buf][row * LD_ + col],
                      &hs[(size_t)(m0 + row) * H_ + kk + col]);
    }
#pragma unroll
    for (int c = 0; c < 4; ++c) {
      int chunk = tid + c * 128;
      int row = chunk >> 3;
      int col = (chunk & 7) * 8;
      async_load_b128((unsigned)(size_t)&Ws[buf][row * LD_ + col],
                      &W[(size_t)(n0 + row) * H_ + kk + col]);
    }
  };

  issue(0, 0);
  int buf = 0;
  for (int kk = 0; kk < H_; kk += 64) {
    if (kk + 64 < H_) { issue(buf ^ 1, kk + 64); wait_async_12(); }
    else              { wait_async_0(); }
    __syncthreads();
#pragma unroll
    for (int kc = 0; kc < 2; ++kc) {
      v16h a0 = load_a_frag(&As[buf][(wave * 32) * LD_ + kc * 32], LD_, lane);
      v16h a1 = load_a_frag(&As[buf][(wave * 32 + 16) * LD_ + kc * 32], LD_, lane);
#pragma unroll
      for (int j = 0; j < 4; ++j) {
        v16h b = load_bT_frag(&Ws[buf][(j * 16) * LD_ + kc * 32], LD_, lane);
        acc[0][j] = wmma_f16(a0, b, acc[0][j]);
        acc[1][j] = wmma_f16(a1, b, acc[1][j]);
      }
    }
    __syncthreads();   // buffer consumed by all waves -> safe to refill
    buf ^= 1;
  }

  if (z < 2) {
    // bias + RoPE in registers: pair (d, d+32) lives in fragments (jp, jp+2).
#pragma unroll
    for (int g = 0; g < 2; ++g) {
#pragma unroll
      for (int jp = 0; jp < 2; ++jp) {
        int n1 = n0 + jp * 16 + (lane & 15);
        int hh = n1 >> 6;
        int d  = n1 & 63;  // < 32 by construction
        float b1 = bias[n1], b2 = bias[n1 + 32];
        float invf = __expf(-(float)d * 0.28782313662425572f);  // ln(1e4)/32
#pragma unroll
        for (int r = 0; r < 8; ++r) {
          int m = m0 + wave * 32 + g * 16 + r + ((lane >> 4) << 3);
          int s = m & (S_ - 1);
          int bb = m >> 11;
          float cs, sn;
          __sincosf((float)s * invf, &sn, &cs);
          float x1 = acc[g][jp][r] + b1;
          float x2 = acc[g][jp + 2][r] + b2;
          size_t base = (((size_t)bb * NH_ + hh) * S_ + s) * HD_;
          dst[base + d]      = (_Float16)(x1 * cs - x2 * sn);
          dst[base + d + 32] = (_Float16)(x2 * cs + x1 * sn);
        }
      }
    }
  } else {
#pragma unroll
    for (int g = 0; g < 2; ++g) {
#pragma unroll
      for (int j = 0; j < 4; ++j) {
        int n1 = n0 + j * 16 + (lane & 15);
        int hh = n1 >> 6;
        int d  = n1 & 63;
        float bv_ = bias[n1];
#pragma unroll
        for (int r = 0; r < 8; ++r) {
          int m = m0 + wave * 32 + g * 16 + r + ((lane >> 4) << 3);
          int s = m & (S_ - 1);
          int bb = m >> 11;
          size_t base = (((size_t)bb * NH_ + hh) * S_ + s) * HD_;
          dst[base + d] = (_Float16)(acc[g][j][r] + bv_);
        }
      }
    }
  }
}

// ---------------------------------------------------------------------------
// Kernel 2: flash attention. Block = 128 thr (4 waves), 64 queries/block,
// whole HD=64 in registers per wave. K tiles async-loaded to LDS, V tiles
// transposed into LDS; both double-buffered. Online softmax via 16-lane
// xor-shuffles matching the C-fragment row mapping.
// ---------------------------------------------------------------------------
__global__ __launch_bounds__(128)
void flash_attn_kernel(const _Float16* __restrict__ Q,
                       const _Float16* __restrict__ K,
                       const _Float16* __restrict__ V,
                       _Float16* __restrict__ O) {
  const int tid = threadIdx.x;
  const int lane = tid & 31;
  const int wave = tid >> 5;
  const int bh = blockIdx.y;                 // 0..31
  const int b = bh >> 4, h = bh & 15;
  const int s0 = blockIdx.x * 64;

  const _Float16* Qb = Q + (size_t)bh * S_ * HD_;
  const _Float16* Kb = K + (size_t)bh * S_ * HD_;
  const _Float16* Vb = V + (size_t)bh * S_ * HD_;

  __shared__ __align__(32) _Float16 Ks[2][64 * LD_];   // [key][d]
  __shared__ __align__(32) _Float16 Vt[2][64 * LD_];   // [d][key]
  __shared__ __align__(32) _Float16 Ps[4 * 16 * LD_];  // per-wave P tile

  // Q fragments (A-layout) for this wave's 16 queries, pre-scaled by 1/8
  const _Float16* qrow =
      Qb + (size_t)(s0 + wave * 16 + (lane & 15)) * HD_ + ((lane >> 4) * 8);
  v16h a0 = cat8(*(const v8h*)qrow, *(const v8h*)(qrow + 16));        // d 0..31
  v16h a1 = cat8(*(const v8h*)(qrow + 32), *(const v8h*)(qrow + 48)); // d 32..63
  const _Float16 qs = (_Float16)0.125f;
#pragma unroll
  for (int i = 0; i < 16; ++i) { a0[i] *= qs; a1[i] *= qs; }

  v8f o[4];
  float mrow[8], lrow[8];
#pragma unroll
  for (int j = 0; j < 4; ++j)
#pragma unroll
    for (int r = 0; r < 8; ++r) o[j][r] = 0.f;
#pragma unroll
  for (int r = 0; r < 8; ++r) { mrow[r] = -1e30f; lrow[r] = 0.f; }

  // K tile: 4 async 16B chunks/thread. V tile: regular loads + transposed
  // ds stores (covered by the compiler's dscnt wait at the next barrier).
  auto issueKV = [&](int buf, int k0) {
#pragma unroll
    for (int c = 0; c < 4; ++c) {
      int chunk = tid + c * 128;
      int row = chunk >> 3;
      int col = (chunk & 7) * 8;
      async_load_b128((unsigned)(size_t)&Ks[buf][row * LD_ + col],
                      &Kb[(size_t)(k0 + row) * HD_ + col]);
      v8h vv = *(const v8h*)&Vb[(size_t)(k0 + row) * HD_ + col];
#pragma unroll
      for (int e = 0; e < 8; ++e) Vt[buf][(col + e) * LD_ + row] = vv[e];
    }
  };

  issueKV(0, 0);
  int buf = 0;
  for (int k0 = 0; k0 < S_; k0 += 64) {
    if (k0 + 64 < S_) { issueKV(buf ^ 1, k0 + 64); wait_async_4(); }
    else              { wait_async_0(); }
    __syncthreads();

    // scores S = Q * K^T : 16 x 64, f32
    v8f s4[4];
#pragma unroll
    for (int j = 0; j < 4; ++j)
#pragma unroll
      for (int r = 0; r < 8; ++r) s4[j][r] = 0.f;
#pragma unroll
    for (int kc = 0; kc < 2; ++kc) {
      v16h aq = kc ? a1 : a0;
#pragma unroll
      for (int j = 0; j < 4; ++j) {
        v16h bk_ = load_bT_frag(&Ks[buf][(j * 16) * LD_ + kc * 32], LD_, lane);
        s4[j] = wmma_f16(aq, bk_, s4[j]);
      }
    }

    // online softmax: row m = r + 8*(lane>>4), spread over 16 lanes x 4 frags
#pragma unroll
    for (int r = 0; r < 8; ++r) {
      float mx = fmaxf(fmaxf(s4[0][r], s4[1][r]), fmaxf(s4[2][r], s4[3][r]));
#pragma unroll
      for (int msk = 1; msk < 16; msk <<= 1)
        mx = fmaxf(mx, __shfl_xor(mx, msk, 32));
      float mnew = fmaxf(mrow[r], mx);
      float corr = __expf(mrow[r] - mnew);
      float psum = 0.f;
#pragma unroll
      for (int j = 0; j < 4; ++j) {
        float p = __expf(s4[j][r] - mnew);
        s4[j][r] = p;
        psum += p;
      }
#pragma unroll
      for (int msk = 1; msk < 16; msk <<= 1)
        psum += __shfl_xor(psum, msk, 32);
      lrow[r] = lrow[r] * corr + psum;
      mrow[r] = mnew;
#pragma unroll
      for (int j = 0; j < 4; ++j) o[j][r] = o[j][r] * corr;
    }

    // C-layout -> A-layout relayout of P through per-wave LDS
    _Float16* Pw = &Ps[wave * 16 * LD_];
#pragma unroll
    for (int j = 0; j < 4; ++j)
#pragma unroll
      for (int r = 0; r < 8; ++r)
        Pw[(r + ((lane >> 4) << 3)) * LD_ + j * 16 + (lane & 15)] =
            (_Float16)s4[j][r];

    // O += P * V
#pragma unroll
    for (int kc = 0; kc < 2; ++kc) {
      v16h pa = load_a_frag(Pw + kc * 32, LD_, lane);
#pragma unroll
      for (int j = 0; j < 4; ++j) {
        v16h bv_ = load_bT_frag(&Vt[buf][(j * 16) * LD_ + kc * 32], LD_, lane);
        o[j] = wmma_f16(pa, bv_, o[j]);
      }
    }
    __syncthreads();   // buffer consumed -> safe to refill next iteration
    buf ^= 1;
  }

  // normalize and write [B,S,H] f16
#pragma unroll
  for (int r = 0; r < 8; ++r) {
    int sq = s0 + wave * 16 + r + ((lane >> 4) << 3);
    float linv = 1.0f / lrow[r];
    size_t base = ((size_t)b * S_ + sq) * H_ + h * HD_;
#pragma unroll
    for (int j = 0; j < 4; ++j) {
      int d = j * 16 + (lane & 15);
      O[base + d] = (_Float16)(o[j][r] * linv);
    }
  }
}

// ---------------------------------------------------------------------------
// Kernel 3: output projection, f32 result to d_out. Same 128x64 block tile,
// 32x64 per wave, async double-buffered staging.
// ---------------------------------------------------------------------------
__global__ __launch_bounds__(128)
void out_proj_kernel(const _Float16* __restrict__ X,
                     const _Float16* __restrict__ Wo,
                     const float* __restrict__ bo,
                     float* __restrict__ out) {
  const int tid = threadIdx.x;
  const int lane = tid & 31;
  const int wave = tid >> 5;
  const int n0 = blockIdx.x * 64;
  const int m0 = blockIdx.y * 128;

  __shared__ __align__(32) _Float16 As[2][128 * LD_];
  __shared__ __align__(32) _Float16 Ws[2][64 * LD_];

  v8f acc[2][4];
#pragma unroll
  for (int g = 0; g < 2; ++g)
#pragma unroll
    for (int j = 0; j < 4; ++j)
#pragma unroll
      for (int r = 0; r < 8; ++r) acc[g][j][r] = 0.f;

  auto issue = [&](int buf, int kk) {
#pragma unroll
    for (int c = 0; c < 8; ++c) {
      int chunk = tid + c * 128;
      int row = chunk >> 3;
      int col = (chunk & 7) * 8;
      async_load_b128((unsigned)(size_t)&As[buf][row * LD_ + col],
                      &X[(size_t)(m0 + row) * H_ + kk + col]);
    }
#pragma unroll
    for (int c = 0; c < 4; ++c) {
      int chunk = tid + c * 128;
      int row = chunk >> 3;
      int col = (chunk & 7) * 8;
      async_load_b128((unsigned)(size_t)&Ws[buf][row * LD_ + col],
                      &Wo[(size_t)(n0 + row) * H_ + kk + col]);
    }
  };

  issue(0, 0);
  int buf = 0;
  for (int kk = 0; kk < H_; kk += 64) {
    if (kk + 64 < H_) { issue(buf ^ 1, kk + 64); wait_async_12(); }
    else              { wait_async_0(); }
    __syncthreads();
#pragma unroll
    for (int kc = 0; kc < 2; ++kc) {
      v16h a0 = load_a_frag(&As[buf][(wave * 32) * LD_ + kc * 32], LD_, lane);
      v16h a1 = load_a_frag(&As[buf][(wave * 32 + 16) * LD_ + kc * 32], LD_, lane);
#pragma unroll
      for (int j = 0; j < 4; ++j) {
        v16h b = load_bT_frag(&Ws[buf][(j * 16) * LD_ + kc * 32], LD_, lane);
        acc[0][j] = wmma_f16(a0, b, acc[0][j]);
        acc[1][j] = wmma_f16(a1, b, acc[1][j]);
      }
    }
    __syncthreads();
    buf ^= 1;
  }

#pragma unroll
  for (int g = 0; g < 2; ++g) {
#pragma unroll
    for (int j = 0; j < 4; ++j) {
      int n1 = n0 + j * 16 + (lane & 15);
      float bv_ = bo[n1];
#pragma unroll
      for (int r = 0; r < 8; ++r) {
        int m = m0 + wave * 32 + g * 16 + r + ((lane >> 4) << 3);
        out[(size_t)m * H_ + n1] = acc[g][j][r] + bv_;
      }
    }
  }
}

// ---------------------------------------------------------------------------
extern "C" void kernel_launch(void* const* d_in, const int* in_sizes, int n_in,
                              void* d_out, int out_size, void* d_ws,
                              size_t ws_size, hipStream_t stream) {
  (void)in_sizes; (void)n_in; (void)out_size; (void)ws_size;
  const float* hs = (const float*)d_in[0];
  const float* Wq = (const float*)d_in[1];
  const float* bq = (const float*)d_in[2];
  const float* Wk = (const float*)d_in[3];
  const float* bk = (const float*)d_in[4];
  const float* Wv = (const float*)d_in[5];
  const float* bv = (const float*)d_in[6];
  const float* Wo = (const float*)d_in[7];
  const float* bo = (const float*)d_in[8];
  float* out = (float*)d_out;

  // workspace layout (f16): ~50.3 MB total
  _Float16* p = (_Float16*)d_ws;
  _Float16* hs16 = p; p += (size_t)M_ * H_;
  _Float16* Wq16 = p; p += (size_t)H_ * H_;
  _Float16* Wk16 = p; p += (size_t)H_ * H_;
  _Float16* Wv16 = p; p += (size_t)H_ * H_;
  _Float16* Wo16 = p; p += (size_t)H_ * H_;
  _Float16* Q16  = p; p += (size_t)M_ * H_;
  _Float16* K16  = p; p += (size_t)M_ * H_;
  _Float16* V16  = p; p += (size_t)M_ * H_;
  _Float16* AO16 = p; p += (size_t)M_ * H_;

  const int nHS4 = M_ * H_ / 4;   // 1,048,576
  const int nW4  = H_ * H_ / 4;   // 262,144
  cast_f32_to_f16<<<(nHS4 + 255) / 256, 256, 0, stream>>>(hs, hs16, nHS4);
  cast_f32_to_f16<<<(nW4 + 255) / 256, 256, 0, stream>>>(Wq, Wq16, nW4);
  cast_f32_to_f16<<<(nW4 + 255) / 256, 256, 0, stream>>>(Wk, Wk16, nW4);
  cast_f32_to_f16<<<(nW4 + 255) / 256, 256, 0, stream>>>(Wv, Wv16, nW4);
  cast_f32_to_f16<<<(nW4 + 255) / 256, 256, 0, stream>>>(Wo, Wo16, nW4);

  qkv_rope_kernel<<<dim3(H_ / 64, M_ / 128, 3), 128, 0, stream>>>(
      hs16, Wq16, Wk16, Wv16, bq, bk, bv, Q16, K16, V16);

  flash_attn_kernel<<<dim3(S_ / 64, B_ * NH_), 128, 0, stream>>>(
      Q16, K16, V16, AO16);

  out_proj_kernel<<<dim3(H_ / 64, M_ / 128), 128, 0, stream>>>(
      AO16, Wo16, bo, out);
}
